// SQConv2d_85684597555510
// MI455X (gfx1250) — 8-bit matrix operand order repaired and checked by reading, compiled, not run on hardware
//
#include <hip/hip_runtime.h>


typedef __attribute__((ext_vector_type(8))) int   v8i;
typedef __attribute__((ext_vector_type(4))) float v4f;

#define ICKK 1152
#define NOUT 256
#define MTOT 65536
#define MT   128
#define KT   64
#define NKC  (ICKK / KT)

__global__ __launch_bounds__(256)
void sq_wscale(const float* __restrict__ w, float* __restrict__ w_scale,
               float* __restrict__ act_scale) {
  int j = blockIdx.x * 256 + threadIdx.x;
  if (j >= ICKK) return;
  float m = 0.f;
  for (int i = 0; i < NOUT; ++i) m = fmaxf(m, fabsf(w[i * ICKK + j]));
  w_scale[j] = m;
  act_scale[j] = 0.f;
}

__global__ __launch_bounds__(256)
void sq_actscale(const float* __restrict__ x, float* __restrict__ act_scale) {
  const int tid = threadIdx.x;
  const int bc  = blockIdx.x;
  const int c   = bc & 127;
  const float* xp = x + (bc << 12);
  float m[9];
#pragma unroll
  for (int k = 0; k < 9; ++k) m[k] = 0.f;
  for (int sp = tid; sp < 4096; sp += 256) {
    float a = fabsf(xp[sp]);
    int h = sp >> 6, w = sp & 63;
    float aw0 = (w <= 62) ? a : 0.f;
    float aw2 = (w >= 1)  ? a : 0.f;
    float b0  = (h <= 62) ? a   : 0.f;
    float c00 = (h <= 62) ? aw0 : 0.f;
    float c02 = (h <= 62) ? aw2 : 0.f;
    float b2  = (h >= 1)  ? a   : 0.f;
    float c20 = (h >= 1)  ? aw0 : 0.f;
    float c22 = (h >= 1)  ? aw2 : 0.f;
    m[0] = fmaxf(m[0], c00); m[1] = fmaxf(m[1], b0);  m[2] = fmaxf(m[2], c02);
    m[3] = fmaxf(m[3], aw0); m[4] = fmaxf(m[4], a);   m[5] = fmaxf(m[5], aw2);
    m[6] = fmaxf(m[6], c20); m[7] = fmaxf(m[7], b2);  m[8] = fmaxf(m[8], c22);
  }
  __shared__ float red[9][256];
#pragma unroll
  for (int k = 0; k < 9; ++k) red[k][tid] = m[k];
  __syncthreads();
  for (int s = 128; s > 0; s >>= 1) {
    if (tid < s) {
#pragma unroll
      for (int k = 0; k < 9; ++k)
        red[k][tid] = fmaxf(red[k][tid], red[k][tid + s]);
    }
    __syncthreads();
  }
  if (tid < 9)
    atomicMax(reinterpret_cast<unsigned*>(&act_scale[c * 9 + tid]),
              __float_as_uint(red[tid][0]));
}

__global__ __launch_bounds__(256)
void sq_scales(const float* __restrict__ w_scale, const float* __restrict__ act_scale,
               float* __restrict__ scale, float* __restrict__ wq_mul,
               float* __restrict__ xq_mul, float* __restrict__ s2) {
  __shared__ float rw[256], rx[256];
  int tid = threadIdx.x;
  float mw = 0.f, mx = 0.f;
  for (int j = tid; j < ICKK; j += 256) {
    float a = act_scale[j], wv = w_scale[j];
    float sc = sqrtf(a) / sqrtf(wv);
    if (!(sc > 0.f && sc < 3.0e38f)) sc = 1.0f;
    scale[j] = sc;
    mw = fmaxf(mw, wv * sc);
    mx = fmaxf(mx, a / sc);
  }
  rw[tid] = mw; rx[tid] = mx;
  __syncthreads();
  for (int s = 128; s > 0; s >>= 1) {
    if (tid < s) {
      rw[tid] = fmaxf(rw[tid], rw[tid + s]);
      rx[tid] = fmaxf(rx[tid], rx[tid + s]);
    }
    __syncthreads();
  }
  float sw = rw[0] * (1.0f / 127.0f); if (sw == 0.f) sw = 1.f;
  float sx = rx[0] * (1.0f / 127.0f); if (sx == 0.f) sx = 1.f;
  if (tid == 0) s2[0] = sw * sx;
  for (int j = tid; j < ICKK; j += 256) {
    float sc = scale[j];
    wq_mul[j] = sc / sw;
    xq_mul[j] = 1.0f / (sc * sx);
  }
}

__global__ __launch_bounds__(256)
void sq_quantw(const float* __restrict__ w, const float* __restrict__ wq_mul,
               signed char* __restrict__ qw) {
  int n = blockIdx.x;
  for (int jp = threadIdx.x; jp < ICKK; jp += 256) {
    int c = jp & 127, khkw = jp >> 7;
    int j = c * 9 + khkw;
    float v = w[n * ICKK + j] * wq_mul[j];
    v = fminf(fmaxf(v, -127.f), 127.f);
    qw[n * ICKK + jp] = (signed char)__float2int_rn(v);
  }
}

__global__ __launch_bounds__(256)
void sq_gemm(const float* __restrict__ x, const signed char* __restrict__ qw,
             const float* __restrict__ xq_mul, const float* __restrict__ s2p,
             const float* __restrict__ bias, float* __restrict__ out) {
  __shared__ __align__(16) signed char Atile[2][MT * KT];
  __shared__ float xs[ICKK];

  const int tid    = threadIdx.x;
  const int blockM = blockIdx.x * MT;
  const int bb     = blockM >> 12;
  const int sp0    = blockM & 4095;

  for (int jp = tid; jp < ICKK; jp += 256) {
    int c = jp & 127, khkw = jp >> 7;
    xs[jp] = xq_mul[c * 9 + khkw];
  }

  const int wave = tid >> 5, lane = tid & 31;
  const int half = lane >> 4, l16 = lane & 15;
  const int mw = wave & 1;
  const int nw = wave >> 1;

  auto produce = [&](int it, signed char* __restrict__ buf) {
    const int khkw = it % 9;
    const int c0   = (it / 9) << 6;
    const int kh   = khkw / 3;
    const int kw   = khkw - 3 * kh;
    const float* xsrow = &xs[khkw * 128 + c0];
#pragma unroll
    for (int itp = 0; itp < 8; ++itp) {
      int d  = tid + itp * 256;
      int rl = d >> 4;
      int cq = (d & 15) << 2;
      int sp = sp0 + rl;
      int oh = sp >> 6, ow = sp & 63;
      int ih = oh + kh - 1, iw = ow + kw - 1;
      unsigned packed = 0;
      if ((unsigned)ih < 64u && (unsigned)iw < 64u) {
        const float* xp = x + (((bb << 7) + c0 + cq) << 12) + (ih << 6) + iw;
#pragma unroll
        for (int u = 0; u < 4; ++u) {
          float v = xp[u << 12] * xsrow[cq + u];
          v = fminf(fmaxf(v, -127.f), 127.f);
          int q = __float2int_rn(v);
          packed |= ((unsigned)q & 0xFFu) << (u * 8);
        }
      }
      *reinterpret_cast<unsigned*>(&buf[d << 2]) = packed;
    }
  };

  v8i acc[4][4];
  const v8i vz = {0, 0, 0, 0, 0, 0, 0, 0};
#pragma unroll
  for (int a = 0; a < 4; ++a)
#pragma unroll
    for (int b = 0; b < 4; ++b) acc[a][b] = vz;

  __syncthreads();
  produce(0, Atile[0]);
  __syncthreads();

  for (int it = 0; it < NKC; ++it) {
    const int khkw = it % 9;
    const int k0 = khkw * 128 + (it / 9) * 64;
    const int itn = it + 1;
    const int kn = (itn < NKC) ? (itn % 9) * 128 + (itn / 9) * 64 : 0;
    const signed char* cur = Atile[it & 1];

    v8i afr[4];
#pragma unroll
    for (int mi = 0; mi < 4; ++mi) {
      int base = (mw * 64 + mi * 16 + l16) * KT + half * 16;
      int2 p0 = *reinterpret_cast<const int2*>(&cur[base]);
      int2 p1 = *reinterpret_cast<const int2*>(&cur[base + 8]);
      int2 p2 = *reinterpret_cast<const int2*>(&cur[base + 32]);
      int2 p3 = *reinterpret_cast<const int2*>(&cur[base + 40]);
      v8i a;
      a[0] = p0.x; a[1] = p0.y; a[2] = p1.x; a[3] = p1.y;
      a[4] = p2.x; a[5] = p2.y; a[6] = p3.x; a[7] = p3.y;
      afr[mi] = a;
    }

    if (itn < NKC) produce(itn, Atile[itn & 1]);

    v8i bfr[4];
#pragma unroll
    for (int ni = 0; ni < 4; ++ni) {
      int n = nw * 64 + ni * 16 + l16;
      const signed char* bp = qw + n * ICKK + k0 + half * 16;
      int4 q0 = *reinterpret_cast<const int4*>(bp);
      int4 q1 = *reinterpret_cast<const int4*>(bp + 32);
      v8i bvec;
      bvec[0] = q0.x; bvec[1] = q0.y; bvec[2] = q0.z; bvec[3] = q0.w;
      bvec[4] = q1.x; bvec[5] = q1.y; bvec[6] = q1.z; bvec[7] = q1.w;
      bfr[ni] = bvec;
      if (itn < NKC)
        __builtin_prefetch(qw + n * ICKK + kn + half * 16, 0, 3);
    }

#pragma unroll
    for (int mi = 0; mi < 4; ++mi)
#pragma unroll
      for (int ni = 0; ni < 4; ++ni)
        acc[mi][ni] = __builtin_amdgcn_wmma_i32_16x16x64_iu8(
            true, afr[mi], true, bfr[ni], acc[mi][ni], false, false);

    __syncthreads();
  }

  const float s2 = s2p[0];
#pragma unroll
  for (int ni = 0; ni < 4; ++ni) {
    int n = nw * 64 + ni * 16 + l16;
    float bv = bias[n];
    float* dst = out + (((bb << 8) + n) << 12);
#pragma unroll
    for (int mi = 0; mi < 4; ++mi) {
      int spb = sp0 + mw * 64 + mi * 16 + half * 8;
      v4f r0, r1;
#pragma unroll
      for (int e = 0; e < 4; ++e) {
        r0[e] = s2 * (float)acc[mi][ni][e] + bv;
        r1[e] = s2 * (float)acc[mi][ni][e + 4] + bv;
      }
      __builtin_nontemporal_store(r0, reinterpret_cast<v4f*>(dst + spb));
      __builtin_nontemporal_store(r1, reinterpret_cast<v4f*>(dst + spb + 4));
    }
  }
}

extern "C" void kernel_launch(void* const* d_in, const int* in_sizes, int n_in,
                              void* d_out, int out_size, void* d_ws, size_t ws_size,
                              hipStream_t stream) {
  (void)in_sizes; (void)n_in; (void)out_size; (void)ws_size;
  const float* x    = (const float*)d_in[0];
  const float* w    = (const float*)d_in[1];
  const float* bias = (const float*)d_in[2];
  float* out = (float*)d_out;

  float* wsf       = (float*)d_ws;
  float* w_scale   = wsf;
  float* act_scale = wsf + ICKK;
  float* scale     = wsf + 2 * ICKK;
  float* wq_mul    = wsf + 3 * ICKK;
  float* xq_mul    = wsf + 4 * ICKK;
  float* s2        = wsf + 5 * ICKK;
  signed char* qw  = (signed char*)d_ws + 32768;

  sq_wscale  <<<(ICKK + 255) / 256, 256, 0, stream>>>(w, w_scale, act_scale);
  sq_actscale<<<16 * 128,          256, 0, stream>>>(x, act_scale);
  sq_scales  <<<1,                 256, 0, stream>>>(w_scale, act_scale, scale,
                                                     wq_mul, xq_mul, s2);
  sq_quantw  <<<NOUT,              256, 0, stream>>>(w, wq_mul, qw);
  sq_gemm    <<<MTOT / MT,         256, 0, stream>>>(x, qw, xq_mul, s2, bias, out);
}
